// ESRA_27324581937602
// MI455X (gfx1250) — matrix operand order repaired and checked by reading, not compiled, not run on hardware
//
#include <hip/hip_runtime.h>
#include <hip/hip_bf16.h>
#include <math.h>

typedef __attribute__((ext_vector_type(16))) _Float16 v16h;
typedef __attribute__((ext_vector_type(8)))  float    v8f;

static constexpr int NB = 16;    // batch
static constexpr int NC = 256;   // channels
static constexpr int NH = 64;
static constexpr int NW = 64;
static constexpr int NR = 16;    // SE reduced dim
static constexpr int HW = NH * NW;

__device__ __forceinline__ float4 f4add(float4 a, float4 b) {
    return make_float4(a.x + b.x, a.y + b.y, a.z + b.z, a.w + b.w);
}
__device__ __forceinline__ float4 f4max(float4 a, float4 b) {
    return make_float4(fmaxf(a.x, b.x), fmaxf(a.y, b.y), fmaxf(a.z, b.z), fmaxf(a.w, b.w));
}
__device__ __forceinline__ float4 f4fma(float s, float4 v, float4 acc) {
    return make_float4(fmaf(s, v.x, acc.x), fmaf(s, v.y, acc.y),
                       fmaf(s, v.z, acc.z), fmaf(s, v.w, acc.w));
}
__device__ __forceinline__ float f4dot(float4 a, float4 b) {
    return a.x * b.x + a.y * b.y + a.z * b.z + a.w * b.w;
}
// softmax over a singleton axis: 1.0 for finite s, NaN-propagating otherwise
__device__ __forceinline__ float softmax1(float s) {
    float e = expf(s - s);
    return e / e;
}

// ---------------------------------------------------------------------------
// Kernel 1: pooled[b][c] = mean_{h,w} x[b,c,h,w].   One block per (b,c) plane.
// ---------------------------------------------------------------------------
__global__ void esra_pool_kernel(const float* __restrict__ x, float* __restrict__ pooled) {
    const int c = blockIdx.x;
    const int b = blockIdx.y;
    const int t = threadIdx.x;  // 256 threads
    const float4* p = (const float4*)(x + (size_t)(b * NC + c) * HW);
    float s = 0.f;
#pragma unroll
    for (int i = 0; i < 4; ++i) {
        float4 v = p[t + i * 256];
        s += (v.x + v.y) + (v.z + v.w);
    }
    __shared__ float sm[256];
    sm[t] = s;
    __syncthreads();
    for (int off = 128; off > 0; off >>= 1) {
        if (t < off) sm[t] += sm[t + off];
        __syncthreads();
    }
    if (t == 0) pooled[b * NC + c] = sm[0] * (1.0f / (float)HW);
}

// ---------------------------------------------------------------------------
// Kernel 2: SE MLP on WMMA (single wave).
//   h1   = relu(pooled[16,256] @ se_w1^T[256,16] + b1)      (8x wmma, K=256)
//   gate = sigmoid(h1[16,16] @ se_w2^T[16,256] + b2)        (16x wmma, K padded 16->32)
//   gw[b,c] = gate[b,c] * conv_w[c]
// ---------------------------------------------------------------------------
__global__ void esra_se_wmma_kernel(const float* __restrict__ pooled,
                                    const float* __restrict__ se_w1,
                                    const float* __restrict__ se_b1,
                                    const float* __restrict__ se_w2,
                                    const float* __restrict__ se_b2,
                                    const float* __restrict__ conv_w,
                                    float* __restrict__ gw) {
    const int lane  = threadIdx.x;   // 32 threads
    const int m     = lane & 15;     // A-row (M) / B-col (N) for this lane
    const int khalf = lane >> 4;     // lane-half selects K sub-range

    __shared__ _Float16 h1[16][16];  // h1[M][K]

    // ---- GEMM1: D[16,16] = pooled[16,256] x se_w1^T[256,16] ----
    v8f acc = {0.f, 0.f, 0.f, 0.f, 0.f, 0.f, 0.f, 0.f};
#pragma unroll 1
    for (int kk = 0; kk < 8; ++kk) {
        const int kb = kk * 32;
        v16h a, bm;
        // A layout (16-bit A 16x32): lanes 0-15 K=0..15 split, lanes 16-31 K shifted by 8;
        // VGPRs 0-3 cover K base, VGPRs 4-7 cover K+16.
#pragma unroll
        for (int i = 0; i < 8; ++i) {
            const int k0 = kb + ((i < 4) ? (khalf * 8 + 2 * i)
                                         : (16 + khalf * 8 + 2 * (i - 4)));
            a[2 * i]     = (_Float16)pooled[m * NC + k0];
            a[2 * i + 1] = (_Float16)pooled[m * NC + k0 + 1];
        }
        // B layout (16-bit B 32x16): lane = N; same K order as A (lanes 16-31 K shifted by 8; elems 8-15 cover K+16).
        // B[k][n] = se_w1[n][k]  (se_w1 is [R=16, C=256], we need its transpose)
        const int kbB = kb + khalf * 8;
#pragma unroll
        for (int j = 0; j < 8; ++j) {
            const int k0 = kbB + ((j < 4) ? 2 * j : (16 + 2 * (j - 4)));
            bm[2 * j]     = (_Float16)se_w1[m * NC + k0];
            bm[2 * j + 1] = (_Float16)se_w1[m * NC + k0 + 1];
        }
        acc = __builtin_amdgcn_wmma_f32_16x16x32_f16(false, a, false, bm,
                                                     (short)0, acc, false, false);
    }
    // D layout: acc[r] -> M = r + 8*khalf, N = lane&15
#pragma unroll
    for (int r = 0; r < 8; ++r) {
        const int M = r + 8 * khalf;
        float v = acc[r] + se_b1[m];
        h1[M][m] = (_Float16)(v > 0.f ? v : 0.f);
    }
    __syncthreads();

    // ---- GEMM2: gate_pre[16,256] = h1[16,16] x se_w2^T[16,256], K padded to 32 ----
    v16h a2;
#pragma unroll
    for (int i = 0; i < 8; ++i) {
        if (i < 4) {
            const int k0 = khalf * 8 + 2 * i;   // 0..15
            a2[2 * i]     = h1[m][k0];
            a2[2 * i + 1] = h1[m][k0 + 1];
        } else {                                 // K >= 16 : zero pad
            a2[2 * i]     = (_Float16)0.f;
            a2[2 * i + 1] = (_Float16)0.f;
        }
    }
#pragma unroll 1
    for (int tile = 0; tile < 16; ++tile) {
        const int cglob = tile * 16 + m;        // this lane's output channel (N)
        v16h b2;
#pragma unroll
        for (int j = 0; j < 8; ++j) {
            const int k0 = (j < 4) ? (khalf * 8 + 2 * j) : (16 + khalf * 8 + 2 * (j - 4));  // K>=16 -> zero pad
            b2[2 * j]     = (k0     < NR) ? (_Float16)se_w2[cglob * NR + k0]     : (_Float16)0.f;
            b2[2 * j + 1] = (k0 + 1 < NR) ? (_Float16)se_w2[cglob * NR + k0 + 1] : (_Float16)0.f;
        }
        v8f z = {0.f, 0.f, 0.f, 0.f, 0.f, 0.f, 0.f, 0.f};
        v8f d = __builtin_amdgcn_wmma_f32_16x16x32_f16(false, a2, false, b2,
                                                       (short)0, z, false, false);
#pragma unroll
        for (int r = 0; r < 8; ++r) {
            const int M = r + 8 * khalf;        // batch index
            float g = d[r] + se_b2[cglob];
            g = 1.0f / (1.0f + expf(-g));       // sigmoid
            gw[M * NC + cglob] = g * conv_w[cglob];
        }
    }
}

// ---------------------------------------------------------------------------
// Kernel 3: per (b,h) row block of 256 threads.
//   x1[w]=mean_c x, x2[w]=max_c x, kv[w]=sum_c gw[b,c]*x + conv_b
//   s1=<x1,kv>*scale, s2=<x1,x2>*scale, s3=<x2,kv>*scale
//   a = softmax1(s1)+softmax1(s2)+softmax1(s3)  (== 3 for finite scores)
//   out[b,c,h,w] = a * x[b,c,h,w]
// ---------------------------------------------------------------------------
__global__ void esra_main_kernel(const float* __restrict__ x,
                                 const float* __restrict__ gw,
                                 const float* __restrict__ conv_b,
                                 float* __restrict__ out) {
    const int h = blockIdx.x;
    const int b = blockIdx.y;
    const int t  = threadIdx.x;   // 256
    const int wq = t & 15;        // float4 column group (w = 4*wq .. 4*wq+3)
    const int cg = t >> 4;        // 0..15 channel group

    const float4* x4 = (const float4*)x;
    float4*       o4 = (float4*)out;
    const size_t rowbase = (size_t)(b * NC) * (HW / 4) + (size_t)h * (NW / 4);

    float4 s    = make_float4(0.f, 0.f, 0.f, 0.f);
    float4 kacc = make_float4(0.f, 0.f, 0.f, 0.f);
    float4 mx   = make_float4(-3.402823466e38f, -3.402823466e38f,
                              -3.402823466e38f, -3.402823466e38f);
#pragma unroll 4
    for (int c = cg; c < NC; c += 16) {
        float4 v = x4[rowbase + (size_t)c * (HW / 4) + wq];
        s    = f4add(s, v);
        mx   = f4max(mx, v);
        kacc = f4fma(gw[b * NC + c], v, kacc);
    }

    __shared__ float4 smS[16][16], smK[16][16], smM[16][16];  // [cg][wq], 12 KiB
    __shared__ float  pbuf[3][16];
    __shared__ float  a_sh;
    smS[cg][wq] = s;
    smK[cg][wq] = kacc;
    smM[cg][wq] = mx;
    __syncthreads();

    if (t < 16) {  // t == wq for cg==0: combine the 16 channel-group partials
        float4 ss = smS[0][t], kk = smK[0][t], mm = smM[0][t];
#pragma unroll
        for (int g = 1; g < 16; ++g) {
            ss = f4add(ss, smS[g][t]);
            kk = f4add(kk, smK[g][t]);
            mm = f4max(mm, smM[g][t]);
        }
        const float cb = conv_b[0];
        float4 x1 = make_float4(ss.x * (1.0f / NC), ss.y * (1.0f / NC),
                                ss.z * (1.0f / NC), ss.w * (1.0f / NC));
        float4 kv = make_float4(kk.x + cb, kk.y + cb, kk.z + cb, kk.w + cb);
        pbuf[0][t] = f4dot(x1, kv);
        pbuf[1][t] = f4dot(x1, mm);
        pbuf[2][t] = f4dot(mm, kv);
    }
    __syncthreads();

    if (t == 0) {
        float s1 = 0.f, s2 = 0.f, s3 = 0.f;
#pragma unroll
        for (int i = 0; i < 16; ++i) { s1 += pbuf[0][i]; s2 += pbuf[1][i]; s3 += pbuf[2][i]; }
        const float scale = 0.125f;   // W^-0.5 = 64^-0.5
        a_sh = softmax1(s1 * scale) + softmax1(s2 * scale) + softmax1(s3 * scale);
    }
    __syncthreads();
    const float a = a_sh;

#pragma unroll 4
    for (int c = cg; c < NC; c += 16) {
        const size_t idx = rowbase + (size_t)c * (HW / 4) + wq;
        float4 v = x4[idx];   // L2-resident re-read
        o4[idx] = make_float4(a * v.x, a * v.y, a * v.z, a * v.w);
    }
}

// ---------------------------------------------------------------------------
extern "C" void kernel_launch(void* const* d_in, const int* in_sizes, int n_in,
                              void* d_out, int out_size, void* d_ws, size_t ws_size,
                              hipStream_t stream) {
    const float* x      = (const float*)d_in[0];
    const float* se_w1  = (const float*)d_in[1];
    const float* se_b1  = (const float*)d_in[2];
    const float* se_w2  = (const float*)d_in[3];
    const float* se_b2  = (const float*)d_in[4];
    const float* conv_w = (const float*)d_in[5];
    const float* conv_b = (const float*)d_in[6];
    float* out = (float*)d_out;

    float* pooled = (float*)d_ws;        // [16, 256]
    float* gw     = pooled + NB * NC;    // [16, 256]

    esra_pool_kernel<<<dim3(NC, NB), 256, 0, stream>>>(x, pooled);
    esra_se_wmma_kernel<<<1, 32, 0, stream>>>(pooled, se_w1, se_b1, se_w2, se_b2,
                                              conv_w, gw);
    esra_main_kernel<<<dim3(NH, NB), 256, 0, stream>>>(x, gw, conv_b, out);
}
